// SimpleMoEBlock_46780783788611
// MI455X (gfx1250) — matrix operand order repaired and checked by reading, not compiled, not run on hardware
//
#include <hip/hip_runtime.h>
#include <math.h>

// ---------------------------------------------------------------------------
// Types for CDNA5 WMMA (wave32): v_wmma_f32_16x16x32_f16
// ---------------------------------------------------------------------------
typedef __attribute__((ext_vector_type(16))) _Float16 v16h;
typedef __attribute__((ext_vector_type(8)))  _Float16 v8h;
typedef __attribute__((ext_vector_type(8)))  float    v8f;

#define WMMA_F16(a, b, c) \
  __builtin_amdgcn_wmma_f32_16x16x32_f16(false, (a), false, (b), (short)0, (c), false, false)

// Problem constants (match reference)
static constexpr int Bn  = 2;
static constexpr int Sn  = 2048;
static constexpr int Hn  = 1024;
static constexpr int Fn  = 4096;
static constexpr int En  = 8;
static constexpr int NHn = 8;     // H/128 heads of dim 128
static constexpr int DHn = 128;
static constexpr int Tn  = Bn * Sn;     // 4096 tokens
static constexpr int H3  = 3 * Hn;      // 3072

// ---------------------------------------------------------------------------
// CDNA5 async global->LDS copy (ch.8: ASYNCcnt-tracked, no VGPR round trip).
// lds_off: LDS byte offset (low 32 bits of the flat shared-aperture address).
// ---------------------------------------------------------------------------
__device__ __forceinline__ void async_load_b128(unsigned lds_off, const void* gptr) {
  asm volatile("global_load_async_to_lds_b128 %0, %1, off"
               :: "v"(lds_off), "v"(gptr) : "memory");
}
__device__ __forceinline__ void wait_async0() {
  asm volatile("s_wait_asynccnt 0x0" ::: "memory");
}
__device__ __forceinline__ unsigned lds_off_of(const void* shared_ptr) {
  return (unsigned)(uintptr_t)shared_ptr;   // flat LDS addr: offset in low 32b
}

// ---------------------------------------------------------------------------
// WMMA tile loaders.
// A matrix 16x32 f16 (ISA 7.12.2): lane m = lane&15, half = lane>>4.
//   element j -> K = half*16 + j (16 contiguous halfs of row m)
// B matrix 32x16 f16 from W stored [N,K] row-major: lane n = lane&15;
//   element j -> K = half*16 + j (16 contiguous halfs of row n).
// C/D 16x16 f32: v[i] -> (m = i + 8*half, n = lane&15).
// ---------------------------------------------------------------------------
__device__ __forceinline__ v16h load_a16x32(const _Float16* __restrict__ base,
                                            int lda, int lane) {
  const int m  = lane & 15;
  const int hs = lane >> 4;
  const _Float16* p = base + (size_t)m * lda + hs * 16;
  v8h lo = *(const v8h*)(p);
  v8h hi = *(const v8h*)(p + 8);
  v16h r;
#pragma unroll
  for (int j = 0; j < 8; ++j) { r[j] = lo[j]; r[j + 8] = hi[j]; }
  return r;
}

// Generic-stride B tile loader (global memory, used in attention QK^T)
__device__ __forceinline__ v16h load_b32x16(const _Float16* __restrict__ base,
                                            int ldb, int lane) {
  const int n  = lane & 15;
  const int hs = lane >> 4;
  const _Float16* p = base + (size_t)n * ldb + hs * 16;
  v8h lo = *(const v8h*)(p);
  v8h hi = *(const v8h*)(p + 8);
  v16h r;
#pragma unroll
  for (int j = 0; j < 8; ++j) { r[j] = lo[j]; r[j + 8] = hi[j]; }
  return r;
}

// ---------------------------------------------------------------------------
// Elementwise helpers
// ---------------------------------------------------------------------------
__global__ void cvt_f32_f16_kernel(const float* __restrict__ in,
                                   _Float16* __restrict__ out, size_t n) {
  size_t i = (size_t)blockIdx.x * blockDim.x + threadIdx.x;
  size_t stride = (size_t)gridDim.x * blockDim.x;
  for (; i < n; i += stride) out[i] = (_Float16)in[i];
}

__global__ void copy_f32_kernel(const float* __restrict__ in,
                                float* __restrict__ out, size_t n) {
  size_t i = (size_t)blockIdx.x * blockDim.x + threadIdx.x;
  size_t stride = (size_t)gridDim.x * blockDim.x;
  for (; i < n; i += stride) out[i] = in[i];
}

// LayerNorm over H, one block per token, f16 output for WMMA consumption.
__global__ __launch_bounds__(256)
void layernorm_f16_kernel(const float* __restrict__ x,
                          const float* __restrict__ g,
                          const float* __restrict__ bta,
                          _Float16* __restrict__ out, int H) {
  const int t = blockIdx.x;
  const float* row = x + (size_t)t * H;
  __shared__ float red[256];
  const int tid = threadIdx.x;

  float s = 0.f;
  for (int i = tid; i < H; i += 256) s += row[i];
  red[tid] = s; __syncthreads();
  for (int off = 128; off > 0; off >>= 1) {
    if (tid < off) red[tid] += red[tid + off];
    __syncthreads();
  }
  const float mean = red[0] / (float)H;
  __syncthreads();

  float s2 = 0.f;
  for (int i = tid; i < H; i += 256) { float d = row[i] - mean; s2 += d * d; }
  red[tid] = s2; __syncthreads();
  for (int off = 128; off > 0; off >>= 1) {
    if (tid < off) red[tid] += red[tid + off];
    __syncthreads();
  }
  const float inv = rsqrtf(red[0] / (float)H + 1e-5f);
  __syncthreads();

  for (int i = tid; i < H; i += 256)
    out[(size_t)t * H + i] = (_Float16)((row[i] - mean) * inv * g[i] + bta[i]);
}

// Router: logits -> softmax(E=8) -> top-2 -> per-expert combine weight [E,T].
__global__ void router_topk_kernel(const _Float16* __restrict__ y,
                                   const float* __restrict__ rw,
                                   float* __restrict__ wcomb, int T, int H) {
  int t = blockIdx.x * blockDim.x + threadIdx.x;
  if (t >= T) return;
  float lg[En];
#pragma unroll
  for (int e = 0; e < En; ++e) lg[e] = 0.f;
  const _Float16* yr = y + (size_t)t * H;
  for (int i = 0; i < H; ++i) {
    float yv = (float)yr[i];
#pragma unroll
    for (int e = 0; e < En; ++e) lg[e] += yv * rw[e * H + i];
  }
  float mx = lg[0];
#pragma unroll
  for (int e = 1; e < En; ++e) mx = fmaxf(mx, lg[e]);
  float se = 0.f;
#pragma unroll
  for (int e = 0; e < En; ++e) { lg[e] = expf(lg[e] - mx); se += lg[e]; }
#pragma unroll
  for (int e = 0; e < En; ++e) lg[e] /= se;
  int i0 = 0;
#pragma unroll
  for (int e = 1; e < En; ++e) if (lg[e] > lg[i0]) i0 = e;
  int i1 = (i0 == 0) ? 1 : 0;
#pragma unroll
  for (int e = 0; e < En; ++e)
    if (e != i0 && lg[e] > lg[i1]) i1 = e;
#pragma unroll
  for (int e = 0; e < En; ++e)
    wcomb[(size_t)e * T + t] = (e == i0 || e == i1) ? lg[e] : 0.f;
}

// ---------------------------------------------------------------------------
// Generic WMMA GEMM:  C[M,N] = A[M,K] @ W[N,K]^T (+bias) with fused epilogues.
// Block = 128 threads (4 waves); waves split M (16 rows each), the 64x32 B
// strip is SHARED: staged once per workgroup into LDS with double-buffered
// CDNA5 async global->LDS copies (overlapped with WMMA on the prior buffer).
// LDS row stride padded to 40 halfs so the 16 reading lanes are bank-
// conflict-free (20*k mod 64 is a permutation). Grid = (N/64, M/64).
// ---------------------------------------------------------------------------
enum GemmMode {
  MODE_F32 = 0,        // f32 store: acc + bias
  MODE_F16 = 1,        // f16 store: acc + bias            (QKV)
  MODE_F32_RES = 2,    // f32 store: acc + bias + resid    (out_proj + x)
  MODE_F16_SILU = 3,   // f16 store: silu(acc + bias)      (fc1)
  MODE_F32_ACC = 4     // f32 rmw:  out += (acc+bias)*wrow[m]  (fc2 combine)
};

static constexpr int BROW = 40;   // padded LDS row stride (halfs)

__global__ __launch_bounds__(128)
void gemm_f16_wmma_kernel(const _Float16* __restrict__ A,
                          const _Float16* __restrict__ Bw,
                          const float* __restrict__ bias,
                          const float* __restrict__ resid,
                          const float* __restrict__ wrow,
                          void* __restrict__ outp,
                          int M, int N, int K,
                          int lda, int ldb, int ldc, int mode) {
  const int lane = threadIdx.x & 31;
  const int wave = threadIdx.x >> 5;
  const int m0 = blockIdx.y * 64 + wave * 16;
  const int n0 = blockIdx.x * 64;

  __shared__ __align__(16) _Float16 bS[2][64 * BROW];

  // Stage the 64x32-half B strip for k0 into bS[buf] (async, 2 chunks/thread).
  auto stage = [&](int buf, int k0) {
#pragma unroll
    for (int it = 0; it < 2; ++it) {
      int c = threadIdx.x + it * 128;      // 256 x 16B chunks
      int row = c >> 2;                     // 0..63
      int col = c & 3;                      // 4 x 16B per row
      const _Float16* g = Bw + (size_t)(n0 + row) * ldb + k0 + col * 8;
      async_load_b128(lds_off_of(&bS[buf][row * BROW + col * 8]), g);
    }
  };

  v8f acc0 = {}, acc1 = {}, acc2 = {}, acc3 = {};
  const _Float16* Abase = A + (size_t)m0 * lda;
  const int hs = lane >> 4;
  const int nc = lane & 15;

  stage(0, 0);
  wait_async0();
  __syncthreads();

  int buf = 0;
  for (int k0 = 0; k0 < K; k0 += 32) {
    if (k0 + 32 < K) stage(buf ^ 1, k0 + 32);     // prefetch next strip

    v16h a = load_a16x32(Abase + k0, lda, lane);
#pragma unroll
    for (int nt = 0; nt < 4; ++nt) {
      const _Float16* p = &bS[buf][(nt * 16 + nc) * BROW + hs * 16];
      v8h lo = *(const v8h*)(p);
      v8h hi = *(const v8h*)(p + 8);
      v16h b;
#pragma unroll
      for (int j = 0; j < 8; ++j) { b[j] = lo[j]; b[j + 8] = hi[j]; }
      switch (nt) {
        case 0: acc0 = WMMA_F16(a, b, acc0); break;
        case 1: acc1 = WMMA_F16(a, b, acc1); break;
        case 2: acc2 = WMMA_F16(a, b, acc2); break;
        case 3: acc3 = WMMA_F16(a, b, acc3); break;
      }
    }

    wait_async0();       // next strip landed
    __syncthreads();     // all waves done reading bS[buf], strip visible
    buf ^= 1;
  }

  v8f accs[4] = {acc0, acc1, acc2, acc3};
#pragma unroll
  for (int nt = 0; nt < 4; ++nt) {
#pragma unroll
    for (int i = 0; i < 8; ++i) {
      const int m = m0 + i + 8 * hs;
      const int n = n0 + nt * 16 + nc;
      float v = accs[nt][i] + (bias ? bias[n] : 0.f);
      const size_t idx = (size_t)m * ldc + n;
      switch (mode) {
        case MODE_F32:      ((float*)outp)[idx] = v; break;
        case MODE_F16:      ((_Float16*)outp)[idx] = (_Float16)v; break;
        case MODE_F32_RES:  ((float*)outp)[idx] = v + resid[idx]; break;
        case MODE_F16_SILU: {
          float sv = v / (1.f + expf(-v));
          ((_Float16*)outp)[idx] = (_Float16)sv;
        } break;
        case MODE_F32_ACC:  ((float*)outp)[idx] += v * wrow[m]; break;
      }
    }
  }
}

// ---------------------------------------------------------------------------
// Flash attention: 1 wave per (b, head, 16-query block). DH=128.
// Per 32-key step: V tile copy is issued FIRST via async global->LDS (overlaps
// with the QK^T WMMAs), online softmax with shfl_xor reductions inside the
// 16-lane halves (C-matrix layout), P relayout via LDS into A-matrix layout,
// then 8 PV WMMAs. Output f16 for out_proj.
// ---------------------------------------------------------------------------
__global__ __launch_bounds__(32)
void flash_attn_kernel(const _Float16* __restrict__ qkv,
                       _Float16* __restrict__ o) {
  const int lane = threadIdx.x & 31;
  const int q0 = blockIdx.x * 16;
  const int bh = blockIdx.y;
  const int b = bh / NHn, hd = bh % NHn;
  const int ld = H3;

  const _Float16* Qb = qkv + ((size_t)(b * Sn + q0)) * ld + hd * DHn;
  const _Float16* Kb = qkv + ((size_t)(b * Sn)) * ld + Hn + hd * DHn;
  const _Float16* Vb = qkv + ((size_t)(b * Sn)) * ld + 2 * Hn + hd * DHn;

  __shared__ __align__(16) float    pS[16 * 32];
  __shared__ __align__(16) _Float16 vS[32 * 128];

  // Q block resident in registers: 4 A tiles covering DH=128
  v16h qa0 = load_a16x32(Qb +  0, ld, lane);
  v16h qa1 = load_a16x32(Qb + 32, ld, lane);
  v16h qa2 = load_a16x32(Qb + 64, ld, lane);
  v16h qa3 = load_a16x32(Qb + 96, ld, lane);

  v8f od[8] = {};
  float mrow[8], lrow[8];
#pragma unroll
  for (int i = 0; i < 8; ++i) { mrow[i] = -1e30f; lrow[i] = 0.f; }

  const float sc = 0.08838834764831845f;  // 1/sqrt(128)
  const int hs = lane >> 4;
  const int nn = lane & 15;

  for (int kc = 0; kc < Sn; kc += 32) {
    // ---- issue async V-tile copy first: 32 keys x 128 halfs = 512 x 16B
    //      lane owns column chunk nn of rows {2i + hs}
#pragma unroll
    for (int i = 0; i < 16; ++i) {
      int row = 2 * i + hs;
      const _Float16* g = Vb + (size_t)(kc + row) * ld + nn * 8;
      async_load_b128(lds_off_of(&vS[row * 128 + nn * 8]), g);
    }

    // ---- scores: S0 = Q @ K[kc..kc+15]^T, S1 = Q @ K[kc+16..kc+31]^T
    v8f s0 = {}, s1 = {};
    {
      v16h bk;
      bk = load_b32x16(Kb + (size_t)kc * ld +  0, ld, lane); s0 = WMMA_F16(qa0, bk, s0);
      bk = load_b32x16(Kb + (size_t)kc * ld + 32, ld, lane); s0 = WMMA_F16(qa1, bk, s0);
      bk = load_b32x16(Kb + (size_t)kc * ld + 64, ld, lane); s0 = WMMA_F16(qa2, bk, s0);
      bk = load_b32x16(Kb + (size_t)kc * ld + 96, ld, lane); s0 = WMMA_F16(qa3, bk, s0);
      bk = load_b32x16(Kb + (size_t)(kc + 16) * ld +  0, ld, lane); s1 = WMMA_F16(qa0, bk, s1);
      bk = load_b32x16(Kb + (size_t)(kc + 16) * ld + 32, ld, lane); s1 = WMMA_F16(qa1, bk, s1);
      bk = load_b32x16(Kb + (size_t)(kc + 16) * ld + 64, ld, lane); s1 = WMMA_F16(qa2, bk, s1);
      bk = load_b32x16(Kb + (size_t)(kc + 16) * ld + 96, ld, lane); s1 = WMMA_F16(qa3, bk, s1);
    }

    // ---- online softmax; C layout: row = i + 8*hs lives in lanes sharing hs
    float alpha[8];
#pragma unroll
    for (int i = 0; i < 8; ++i) {
      float a0 = s0[i] * sc, a1 = s1[i] * sc;
      float mx = fmaxf(a0, a1);
#pragma unroll
      for (int msk = 1; msk < 16; msk <<= 1)
        mx = fmaxf(mx, __shfl_xor(mx, msk, 32));
      float mnew = fmaxf(mrow[i], mx);
      float al = expf(mrow[i] - mnew);
      float p0 = expf(a0 - mnew);
      float p1 = expf(a1 - mnew);
      float rs = p0 + p1;
#pragma unroll
      for (int msk = 1; msk < 16; msk <<= 1)
        rs += __shfl_xor(rs, msk, 32);
      lrow[i] = lrow[i] * al + rs;
      mrow[i] = mnew;
      alpha[i] = al;
      pS[(i + 8 * hs) * 32 + nn]      = p0;  // keys kc + nn
      pS[(i + 8 * hs) * 32 + 16 + nn] = p1;  // keys kc + 16 + nn
    }
    wait_async0();       // V tile landed in LDS
    __syncthreads();     // pS stores visible across the wave halves

    // rescale running output
#pragma unroll
    for (int dt = 0; dt < 8; ++dt)
#pragma unroll
      for (int i = 0; i < 8; ++i) od[dt][i] *= alpha[i];

    // P (16x32) in A layout
    v16h pa;
#pragma unroll
    for (int j = 0; j < 16; ++j) {
      int kk = hs * 16 + j;
      pa[j] = (_Float16)pS[nn * 32 + kk];
    }

    // ---- PV: 8 d-tiles of 16 columns
#pragma unroll
    for (int dt = 0; dt < 8; ++dt) {
      v16h vb;
#pragma unroll
      for (int j = 0; j < 16; ++j)
        vb[j] = vS[(hs * 16 + j) * 128 + dt * 16 + nn];
      od[dt] = WMMA_F16(pa, vb, od[dt]);
    }
    __syncthreads();
  }

  // ---- finalize: divide by l, store f16 (head-concat layout [T,H])
#pragma unroll
  for (int dt = 0; dt < 8; ++dt)
#pragma unroll
    for (int i = 0; i < 8; ++i) {
      int q = q0 + i + 8 * hs;
      float val = od[dt][i] / lrow[i];
      o[((size_t)(b * Sn + q)) * Hn + hd * DHn + dt * 16 + nn] = (_Float16)val;
    }
}

// ---------------------------------------------------------------------------
// Host orchestration
// ---------------------------------------------------------------------------
extern "C" void kernel_launch(void* const* d_in, const int* in_sizes, int n_in,
                              void* d_out, int out_size, void* d_ws, size_t ws_size,
                              hipStream_t stream) {
  (void)in_sizes; (void)n_in; (void)out_size;

  const float* x          = (const float*)d_in[0];
  const float* ln_attn_g  = (const float*)d_in[1];
  const float* ln_attn_b  = (const float*)d_in[2];
  const float* ln_mlp_g   = (const float*)d_in[3];
  const float* ln_mlp_b   = (const float*)d_in[4];
  const float* in_proj_w  = (const float*)d_in[5];
  const float* in_proj_b  = (const float*)d_in[6];
  const float* out_proj_w = (const float*)d_in[7];
  const float* out_proj_b = (const float*)d_in[8];
  const float* router_w   = (const float*)d_in[9];
  const float* fc1_w      = (const float*)d_in[10];
  const float* fc1_b      = (const float*)d_in[11];
  const float* fc2_w      = (const float*)d_in[12];
  const float* fc2_b      = (const float*)d_in[13];
  float* out = (float*)d_out;

  // workspace partition (all 256B-aligned)
  uint8_t* ws = (uint8_t*)d_ws;
  size_t off = 0;
  auto take = [&](size_t bytes) -> void* {
    void* p = ws + off;
    off = (off + bytes + 255) & ~(size_t)255;
    return p;
  };
  _Float16* xa   = (_Float16*)take((size_t)Tn * Hn * 2);   // LN1 out
  _Float16* wq   = (_Float16*)take((size_t)H3 * Hn * 2);   // in_proj f16
  _Float16* qkvh = (_Float16*)take((size_t)Tn * H3 * 2);   // qkv f16
  _Float16* ob   = (_Float16*)take((size_t)Tn * Hn * 2);   // attn out f16
  _Float16* wo   = (_Float16*)take((size_t)Hn * Hn * 2);   // out_proj f16
  float*    h    = (float*)   take((size_t)Tn * Hn * 4);   // residual stream
  _Float16* y    = (_Float16*)take((size_t)Tn * Hn * 2);   // LN2 out f16
  float*    wcmb = (float*)   take((size_t)En * Tn * 4);   // [E,T] combine w
  _Float16* wb1  = (_Float16*)take((size_t)Fn * Hn * 2);   // fc1[e] f16
  _Float16* wb2  = (_Float16*)take((size_t)Hn * Fn * 2);   // fc2[e] f16
  _Float16* he   = (_Float16*)take((size_t)Tn * Fn * 2);   // expert hidden f16
  (void)ws_size;

  // 1) LN(attn) -> f16
  layernorm_f16_kernel<<<Tn, 256, 0, stream>>>(x, ln_attn_g, ln_attn_b, xa, Hn);

  // 2) weight conversions
  cvt_f32_f16_kernel<<<2048, 256, 0, stream>>>(in_proj_w, wq, (size_t)H3 * Hn);
  cvt_f32_f16_kernel<<<1024, 256, 0, stream>>>(out_proj_w, wo, (size_t)Hn * Hn);

  // 3) QKV = xa @ in_proj_w^T + b  (f16 out for attention)
  gemm_f16_wmma_kernel<<<dim3(H3 / 64, Tn / 64), 128, 0, stream>>>(
      xa, wq, in_proj_b, nullptr, nullptr, qkvh,
      Tn, H3, Hn, Hn, Hn, H3, MODE_F16);

  // 4) flash attention (B*NH heads, S/16 query blocks)
  flash_attn_kernel<<<dim3(Sn / 16, Bn * NHn), 32, 0, stream>>>(qkvh, ob);

  // 5) h = x + attn @ out_proj_w^T + b
  gemm_f16_wmma_kernel<<<dim3(Hn / 64, Tn / 64), 128, 0, stream>>>(
      ob, wo, out_proj_b, x, nullptr, h,
      Tn, Hn, Hn, Hn, Hn, Hn, MODE_F32_RES);

  // 6) LN(mlp) -> f16
  layernorm_f16_kernel<<<Tn, 256, 0, stream>>>(h, ln_mlp_g, ln_mlp_b, y, Hn);

  // 7) router -> per-expert combine weights [E,T]
  router_topk_kernel<<<(Tn + 127) / 128, 128, 0, stream>>>(y, router_w, wcmb, Tn, Hn);

  // 8) out = h (then accumulate expert contributions)
  copy_f32_kernel<<<4096, 256, 0, stream>>>(h, out, (size_t)Tn * Hn);

  // 9) experts: out += silu(y@fc1^T+b1)@fc2^T+b2, scaled by router weight
  for (int e = 0; e < En; ++e) {
    cvt_f32_f16_kernel<<<4096, 256, 0, stream>>>(
        fc1_w + (size_t)e * Fn * Hn, wb1, (size_t)Fn * Hn);
    gemm_f16_wmma_kernel<<<dim3(Fn / 64, Tn / 64), 128, 0, stream>>>(
        y, wb1, fc1_b + (size_t)e * Fn, nullptr, nullptr, he,
        Tn, Fn, Hn, Hn, Hn, Fn, MODE_F16_SILU);
    cvt_f32_f16_kernel<<<4096, 256, 0, stream>>>(
        fc2_w + (size_t)e * Hn * Fn, wb2, (size_t)Hn * Fn);
    gemm_f16_wmma_kernel<<<dim3(Hn / 64, Tn / 64), 128, 0, stream>>>(
        he, wb2, fc2_b + (size_t)e * Hn, nullptr, wcmb + (size_t)e * Tn, out,
        Tn, Hn, Fn, Fn, Fn, Hn, MODE_F32_ACC);
  }
}
